// SRC_44152263803172
// MI455X (gfx1250) — matrix operand order repaired and checked by reading, not compiled, not run on hardware
//
#include <hip/hip_runtime.h>
#include <hip/hip_bf16.h>

// ---- problem constants (from reference) ----
#define SEQ_LEN 1024
#define BATCH   64
#define INPUT   256
#define HIDDEN  512

#define ALPHA   0.9f
#define RHO     3.0f
#define R_      2.0f
#define RS_     -7.0f
#define Z_      0.0f
#define ZHYP_S  0.9f
#define ZDEP_S  0.0f
#define BH_MAX  -4.0f

typedef __attribute__((ext_vector_type(16))) __bf16 v16bf;
typedef __attribute__((ext_vector_type(8)))  __bf16 v8bf;
typedef __attribute__((ext_vector_type(8)))  float  v8f;

// ---------------------------------------------------------------------------
// Branch-free tanh using hardware transcendentals (v_exp_f32 / v_rcp_f32).
// tanh(x) = sign(x) * (1 - e) / (1 + e),  e = 2^(-2*log2(e)*|x|) = exp(-2|x|).
// No EXEC-divergent regions on the serial scan chain; |err| ~ 1e-6.
// ---------------------------------------------------------------------------
__device__ __forceinline__ float fast_tanh(float x) {
    const float ax = __builtin_fabsf(x);
    const float e  = __builtin_amdgcn_exp2f(ax * (-2.0f * 1.44269504088896340736f));
    const float r  = (1.0f - e) * __builtin_amdgcn_rcpf(1.0f + e);
    return __builtin_copysignf(r, x);
}

// ---------------------------------------------------------------------------
// Elementwise recurrence step (shared by both fused kernels).
// ---------------------------------------------------------------------------
__device__ __forceinline__ void scan_update(float delta, float bhc,
                                            float& vi, float& vh, float& vhs,
                                            float& sout_v) {
    const float ival   = ALPHA * vi + delta;
    const float x      = RHO * fast_tanh(ival * (1.0f / RHO));
    const float hprev  = vh;
    const float hsprev = vhs;
    const float Hbin   = (hprev > 0.5f) ? 1.0f : 0.0f;
    const float zs     = ZHYP_S + (ZDEP_S - ZHYP_S) * Hbin;
    const float hcand  = fast_tanh(x + R_ * hprev + RS_ * hsprev + bhc);
    const float hnew   = Z_ * hprev + (1.0f - Z_) * hcand;
    const float hsnew  = zs * hsprev + (1.0f - zs) * hprev;
    vi  = ival;
    vh  = hnew;
    vhs = hsnew;
    sout_v = fmaxf(hnew, 0.0f);   // fwd of straight-through ReLU
}

// ---------------------------------------------------------------------------
// One-shot pre-pass: sin fp32 (S,B,I) -> bf16 row-major into d_ws.
// Shrinks the L2-resident A stream 67MB -> 33.5MB and removes all cvt work
// from the 1024-step serial loop.
// ---------------------------------------------------------------------------
__global__ __launch_bounds__(256)
void SRC_44152263803172_cvt_bf16(const float* __restrict__ src,
                                 __bf16* __restrict__ dst) {
    const size_t i0 = ((size_t)blockIdx.x * 256 + threadIdx.x) * 8;
    const float4 a = *(const float4*)(src + i0);
    const float4 b = *(const float4*)(src + i0 + 4);
    v8bf o;
    o[0] = (__bf16)a.x; o[1] = (__bf16)a.y; o[2] = (__bf16)a.z; o[3] = (__bf16)a.w;
    o[4] = (__bf16)b.x; o[5] = (__bf16)b.y; o[6] = (__bf16)b.z; o[7] = (__bf16)b.w;
    *(v8bf*)(dst + i0) = o;
}

// ---- A-tile fragment load (ISA 16-bit 16x32 A layout) -----------------------
__device__ __forceinline__ void load_tile(const __bf16* __restrict__ aRow,
                                          int koffA, v16bf afr[8]) {
    #pragma unroll
    for (int kk = 0; kk < 8; ++kk) {
        const __bf16* p0 = aRow + kk * 32 + koffA;       // 16B aligned
        const v8bf lo  = *(const v8bf*)p0;               // K +0..7   (or +8..15)
        const v8bf hic = *(const v8bf*)(p0 + 16);        // K +16..23 (or +24..31)
        afr[kk] = __builtin_shufflevector(lo, hic,
            0, 1, 2, 3, 4, 5, 6, 7, 8, 9, 10, 11, 12, 13, 14, 15);
    }
}

// ---- 16x16 delta tile: 8 WMMAs, two independent accumulator chains ----------
__device__ __forceinline__ v8f gemm_tile(const v16bf afr[8], const v16bf bfrag[8]) {
    v8f acc0 = {}, acc1 = {};
    #pragma unroll
    for (int kk = 0; kk < 8; ++kk) {
        if (kk & 1)
            acc1 = __builtin_amdgcn_wmma_f32_16x16x32_bf16(
                       false, afr[kk], false, bfrag[kk], (short)0, acc1, false, false);
        else
            acc0 = __builtin_amdgcn_wmma_f32_16x16x32_bf16(
                       false, afr[kk], false, bfrag[kk], (short)0, acc0, false, false);
    }
    return acc0 + acc1;
}

// ---------------------------------------------------------------------------
// Fused GEMM (bf16 WMMA, f32 acc) + recurrent scan; bf16 A path.
// One wave32 per 16x16 state tile; grid (HIDDEN/16, BATCH/16) = 128 waves.
// Ws tile pinned in 64 VGPRs; A tiles double-buffered across steps so the
// next step's 16 b128 loads fly while this step's WMMAs + scan execute.
// ---------------------------------------------------------------------------
__global__ __launch_bounds__(32)
void SRC_44152263803172_fused_bf16(const __bf16* __restrict__ sin_bf,  // (S,B,I) bf16
                                   const float* __restrict__ ws_p,     // (H,I)  f32
                                   const float* __restrict__ bh_p,     // (H,)   f32
                                   float* __restrict__ out_p)          // sout ++ i,h,hs
{
    const int lane = threadIdx.x;
    const int n    = lane & 15;
    const int hi   = lane >> 4;
    const int h0   = blockIdx.x * 16;
    const int b0   = blockIdx.y * 16;

    // B fragments: ISA 32x16 16-bit layout (lanes 0-15: K 0..7,16..23; 16-31: K 8..15,24..31)
    v16bf bfrag[8];
    {
        const float* wrow = ws_p + (size_t)(h0 + n) * INPUT + (hi ? 8 : 0);
        #pragma unroll
        for (int kk = 0; kk < 8; ++kk) {
            const float* p = wrow + kk * 32;
            v16bf bb;
            #pragma unroll
            for (int t = 0; t < 16; ++t) bb[t] = (__bf16)p[t < 8 ? t : t + 8];
            bfrag[kk] = bb;
        }
    }

    const float bhc = fminf(bh_p[h0 + n], BH_MAX);
    v8f vi = {}; v8f vh = {}; v8f vhs = {};

    const int rowA  = b0 + n;
    const int koffA = hi ? 8 : 0;
    const __bf16* aBase = sin_bf + (size_t)rowA * INPUT;
    const size_t  aStep = (size_t)BATCH * INPUT;          // elems per time step

    // scan+store body for one step, delta in WMMA C/D register layout
    auto do_step = [&](int s, const v8f& delta) {
        float* soutCol = out_p + ((size_t)s * BATCH + b0) * HIDDEN + (h0 + n);
        #pragma unroll
        for (int v = 0; v < 8; ++v) {
            const int m = v + (hi ? 8 : 0);
            float iv = vi[v], hv = vh[v], hsv = vhs[v], so;
            scan_update(delta[v], bhc, iv, hv, hsv, so);
            vi[v] = iv; vh[v] = hv; vhs[v] = hsv;
            __builtin_nontemporal_store(so, soutCol + (size_t)m * HIDDEN);
        }
    };

    v16bf bufA[8], bufB[8];
    load_tile(aBase, koffA, bufA);                         // step 0

    for (int s = 0; s < SEQ_LEN; s += 2) {
        // --- even step: compute bufA, fetch s+1 into bufB ---
        load_tile(aBase + (size_t)(s + 1) * aStep, koffA, bufB);
        __builtin_prefetch(aBase + (size_t)(s + 2) * aStep, 0, 1);
        do_step(s, gemm_tile(bufA, bfrag));

        // --- odd step: compute bufB, fetch s+2 into bufA ---
        if (s + 2 < SEQ_LEN)
            load_tile(aBase + (size_t)(s + 2) * aStep, koffA, bufA);
        __builtin_prefetch(aBase + (size_t)(s + 3) * aStep, 0, 1);
        do_step(s + 1, gemm_tile(bufB, bfrag));
    }

    float* fs = out_p + (size_t)SEQ_LEN * BATCH * HIDDEN;
    const size_t BH = (size_t)BATCH * HIDDEN;
    #pragma unroll
    for (int v = 0; v < 8; ++v) {
        const int    m   = v + (hi ? 8 : 0);
        const size_t idx = (size_t)(b0 + m) * HIDDEN + (h0 + n);
        fs[idx]          = vi[v];
        fs[BH + idx]     = vh[v];
        fs[2 * BH + idx] = vhs[v];
    }
}

// ---------------------------------------------------------------------------
// Fallback (ws too small): converts fp32 A in-loop, same math.
// ---------------------------------------------------------------------------
__global__ __launch_bounds__(32)
void SRC_44152263803172_fused_f32(const float* __restrict__ sin_p,
                                  const float* __restrict__ ws_p,
                                  const float* __restrict__ bh_p,
                                  float* __restrict__ out_p)
{
    const int lane = threadIdx.x;
    const int n    = lane & 15;
    const int hi   = lane >> 4;
    const int h0   = blockIdx.x * 16;
    const int b0   = blockIdx.y * 16;

    v16bf bfrag[8];
    {
        const float* wrow = ws_p + (size_t)(h0 + n) * INPUT + (hi ? 8 : 0);
        #pragma unroll
        for (int kk = 0; kk < 8; ++kk) {
            const float* p = wrow + kk * 32;
            v16bf bb;
            #pragma unroll
            for (int t = 0; t < 16; ++t) bb[t] = (__bf16)p[t < 8 ? t : t + 8];
            bfrag[kk] = bb;
        }
    }

    const float bhc = fminf(bh_p[h0 + n], BH_MAX);
    v8f vi = {}; v8f vh = {}; v8f vhs = {};

    const int rowA  = b0 + n;
    const int koffA = hi ? 8 : 0;
    const float* aBase = sin_p + (size_t)rowA * INPUT;

    for (int s = 0; s < SEQ_LEN; ++s) {
        const float* aRow = aBase + (size_t)s * BATCH * INPUT;
        if (s + 1 < SEQ_LEN)
            __builtin_prefetch(aRow + (size_t)BATCH * INPUT, 0, 1);

        v8f acc0 = {}, acc1 = {};
        #pragma unroll
        for (int kk = 0; kk < 8; ++kk) {
            const float* p0 = aRow + kk * 32 + koffA;
            const float* p1 = p0 + 16;
            v16bf aa;
            #pragma unroll
            for (int t = 0; t < 8; ++t) {
                aa[t]     = (__bf16)p0[t];
                aa[8 + t] = (__bf16)p1[t];
            }
            if (kk & 1)
                acc1 = __builtin_amdgcn_wmma_f32_16x16x32_bf16(
                           false, aa, false, bfrag[kk], (short)0, acc1, false, false);
            else
                acc0 = __builtin_amdgcn_wmma_f32_16x16x32_bf16(
                           false, aa, false, bfrag[kk], (short)0, acc0, false, false);
        }
        const v8f delta = acc0 + acc1;

        float* soutCol = out_p + ((size_t)s * BATCH + b0) * HIDDEN + (h0 + n);
        #pragma unroll
        for (int v = 0; v < 8; ++v) {
            const int m = v + (hi ? 8 : 0);
            float iv = vi[v], hv = vh[v], hsv = vhs[v], so;
            scan_update(delta[v], bhc, iv, hv, hsv, so);
            vi[v] = iv; vh[v] = hv; vhs[v] = hsv;
            __builtin_nontemporal_store(so, soutCol + (size_t)m * HIDDEN);
        }
    }

    float* fs = out_p + (size_t)SEQ_LEN * BATCH * HIDDEN;
    const size_t BH = (size_t)BATCH * HIDDEN;
    #pragma unroll
    for (int v = 0; v < 8; ++v) {
        const int    m   = v + (hi ? 8 : 0);
        const size_t idx = (size_t)(b0 + m) * HIDDEN + (h0 + n);
        fs[idx]          = vi[v];
        fs[BH + idx]     = vh[v];
        fs[2 * BH + idx] = vhs[v];
    }
}

extern "C" void kernel_launch(void* const* d_in, const int* in_sizes, int n_in,
                              void* d_out, int out_size, void* d_ws, size_t ws_size,
                              hipStream_t stream) {
    (void)in_sizes; (void)n_in; (void)out_size;
    const float* sin_p = (const float*)d_in[0];   // (S,B,I) f32
    const float* ws_p  = (const float*)d_in[1];   // (H,I)  f32
    const float* bh_p  = (const float*)d_in[2];   // (H,)   f32
    float*       out_p = (float*)d_out;

    const size_t nSin      = (size_t)SEQ_LEN * BATCH * INPUT;   // 16,777,216 elems
    const size_t needBytes = nSin * sizeof(__bf16);             // 33.5 MB

    dim3 grid(HIDDEN / 16, BATCH / 16);   // 32 x 4 = 128 single-wave workgroups
    dim3 block(32);

    if (ws_size >= needBytes) {
        __bf16* sin_bf = (__bf16*)d_ws;
        SRC_44152263803172_cvt_bf16<<<dim3((unsigned)(nSin / 8 / 256)), dim3(256), 0, stream>>>(
            sin_p, sin_bf);
        SRC_44152263803172_fused_bf16<<<grid, block, 0, stream>>>(sin_bf, ws_p, bh_p, out_p);
    } else {
        SRC_44152263803172_fused_f32<<<grid, block, 0, stream>>>(sin_p, ws_p, bh_p, out_p);
    }
}
